// NonLocal_6107443495423
// MI455X (gfx1250) — matrix operand order repaired and checked by reading, not compiled, not run on hardware
//
#include <hip/hip_runtime.h>

typedef __attribute__((ext_vector_type(16))) __bf16        v16bf;
typedef __attribute__((ext_vector_type(8)))  __bf16        v8bf;
typedef __attribute__((ext_vector_type(4)))  __bf16        v4bf;
typedef __attribute__((ext_vector_type(8)))  float         v8f;
typedef __attribute__((ext_vector_type(4)))  unsigned int  v4u;
typedef __attribute__((ext_vector_type(4)))  float         v4f;

__device__ __forceinline__ unsigned short f2bf(float f) {
    __bf16 h = (__bf16)f;                       // native v_cvt on gfx1250
    return __builtin_bit_cast(unsigned short, h);
}
__device__ __forceinline__ float bf2f(unsigned short u) {
    return (float)__builtin_bit_cast(__bf16, u);
}

// CDNA5 async copy: 16 bytes global -> LDS per lane, tracked by ASYNCcnt.
// VDST carries the LDS byte offset (low 32 bits of the flat LDS address).
__device__ __forceinline__ void async_copy_b128(void* lds_dst, const void* gsrc) {
    unsigned lds_off = (unsigned)(unsigned long long)lds_dst;
    asm volatile("global_load_async_to_lds_b128 %0, %1, off"
                 :: "v"(lds_off), "v"(gsrc) : "memory");
}
__device__ __forceinline__ void wait_async() {
    asm volatile("s_wait_asynccnt 0x0" ::: "memory");
}

union FragAB { v4u q[2]; v16bf bf; };
static_assert(sizeof(FragAB) == 32, "frag size");

#define BM 128
#define BN 128
#define BK 32

// C[M x N] = A[M x K] * B[K x N], fp32 accumulate via v_wmma_f32_16x16x32_bf16.
// A: row-major [M][K] (fp32 if A_F32, else bf16).
// B: if B_F32  -> fp32 row-major [K][N], ldb = row stride (transposed while staging)
//    if !B_F32 -> bf16 B-TRANSPOSED [N][K], ldb = row stride (direct async copy)
// MODE: 0 = fp32 store C[m*ldo+n]
//       1 = bf16 store C[m*ldo+n]
//       2 = bf16 transposed PACKED store C[n*ldo + m..m+7] (one b128 per tile-row-octet)
//       4 = like 2, but adds residual read packed from aux[n*ldadd + m..m+7]
//       5 = mode 1 store to C  +  mode 2 packed store to aux (dual-layout output)
template<bool A_F32, bool B_F32, int MODE>
__global__ __launch_bounds__(256) void wmma_gemm(
    const void* __restrict__ Ag, const void* __restrict__ Bg,
    void* __restrict__ Cg, unsigned short* __restrict__ auxPtr,
    int Kdim, int lda, int ldb, int ldo, int ldadd,
    long long strideA, long long strideB, long long strideC, long long strideAux)
{
    __shared__ unsigned short ldsA[2][BM * BK];   // A tile, row-major [m][k], double buffered
    __shared__ unsigned short ldsB[2][BN * BK];   // B tile transposed, [n][k]

    const int tid  = threadIdx.x;
    const int lane = tid & 31;
    const int wave = tid >> 5;
    const int wm   = wave & 3;    // 4 wave-rows  -> each covers 2 M-tiles (32 rows)
    const int wn   = wave >> 2;   // 2 wave-cols  -> each covers 4 N-tiles (64 cols)
    const int bz   = blockIdx.z;
    const unsigned m0 = blockIdx.y * BM;
    const unsigned n0 = blockIdx.x * BN;

    const float*          Af = A_F32 ? ((const float*)Ag + (size_t)bz * strideA) : nullptr;
    const unsigned short* Ah = A_F32 ? nullptr : ((const unsigned short*)Ag + (size_t)bz * strideA);
    const float*          Bf = B_F32 ? ((const float*)Bg + (size_t)bz * strideB) : nullptr;
    const unsigned short* Bh = B_F32 ? nullptr : ((const unsigned short*)Bg + (size_t)bz * strideB);

    auto stage = [&](int kk, int buf) {
        if (A_F32) {
            v4f f[4];                               // issue all loads, then convert+store
#pragma unroll
            for (int it = 0; it < 4; ++it) {
                int c   = tid + it * 256;            // 1024 chunks of 4 floats
                int row = c >> 3, seg = c & 7;
                f[it] = *(const v4f*)(Af + (unsigned)(m0 + row) * lda + kk + seg * 4);
            }
#pragma unroll
            for (int it = 0; it < 4; ++it) {
                int c   = tid + it * 256;
                int row = c >> 3, seg = c & 7;
                v4bf h = { (__bf16)f[it].x, (__bf16)f[it].y,   // -> v_cvt_pk_bf16_f32 x2
                           (__bf16)f[it].z, (__bf16)f[it].w };
                *(v4bf*)&ldsA[buf][row * BK + seg * 4] = h;    // ds_store_b64
            }
        } else {
#pragma unroll
            for (int it = 0; it < 2; ++it) {
                int c   = tid + it * 256;            // 512 chunks of 8 bf16
                int row = c >> 2, seg = c & 3;
                const unsigned short* src = Ah + (unsigned)(m0 + row) * lda + kk + seg * 8;
                async_copy_b128(&ldsA[buf][row * BK + seg * 8], src);
                if (kk + BK < Kdim) __builtin_prefetch(src + BK, 0, 0);
            }
        }
        if (B_F32) {
            v4f g[4];
#pragma unroll
            for (int it = 0; it < 4; ++it) {         // fp32 [K][N]: batch loads first
                int c = tid + it * 256;              // 1024 chunks of 4 floats
                int k = c >> 5, seg = c & 31;
                g[it] = *(const v4f*)(Bf + (unsigned)(kk + k) * ldb + n0 + seg * 4);
            }
#pragma unroll
            for (int it = 0; it < 4; ++it) {         // transpose + convert scatter
                int c = tid + it * 256;
                int k = c >> 5, seg = c & 31;
                int nb = seg * 4;
                ldsB[buf][(nb + 0) * BK + k] = f2bf(g[it].x);
                ldsB[buf][(nb + 1) * BK + k] = f2bf(g[it].y);
                ldsB[buf][(nb + 2) * BK + k] = f2bf(g[it].z);
                ldsB[buf][(nb + 3) * BK + k] = f2bf(g[it].w);
            }
        } else {
#pragma unroll
            for (int it = 0; it < 2; ++it) {         // bf16 BT [N][K]: direct async copy
                int c   = tid + it * 256;            // 512 chunks of 8 bf16
                int row = c >> 2, seg = c & 3;
                const unsigned short* src = Bh + (unsigned)(n0 + row) * ldb + kk + seg * 8;
                async_copy_b128(&ldsB[buf][row * BK + seg * 8], src);
                if (kk + BK < Kdim) __builtin_prefetch(src + BK, 0, 0);
            }
        }
    };

    const v8f vzero = {0.f, 0.f, 0.f, 0.f, 0.f, 0.f, 0.f, 0.f};
    v8f acc[2][4];
#pragma unroll
    for (int mt = 0; mt < 2; ++mt)
#pragma unroll
        for (int nt = 0; nt < 4; ++nt)
            acc[mt][nt] = vzero;

    const int nIter = Kdim / BK;
    stage(0, 0);
    if (!A_F32 || !B_F32) wait_async();
    __syncthreads();

    const int khalf = lane >> 4;     // lane half selects K sub-range
    const int rl    = lane & 15;     // row (A) / col (B) within 16x16 tile

    for (int i = 0; i < nIter; ++i) {
        const int cur = i & 1;
        if (i + 1 < nIter) stage((i + 1) * BK, cur ^ 1);   // prefetch next tile (async)

        // ---- build fragments per ISA 7.12.2 layouts and issue WMMA ----
        const unsigned short* lA = ldsA[cur];
        const unsigned short* lB = ldsB[cur];
        FragAB a[2], b[4];
#pragma unroll
        for (int nt = 0; nt < 4; ++nt) { // B: K quarters split lane-halves / upper VGPRs (same as A)
            int n = (wn * 4 + nt) * 16 + rl;
            int base = n * BK + khalf * 8;
            b[nt].q[0] = *(const v4u*)&lB[base];
            b[nt].q[1] = *(const v4u*)&lB[base + 16];
        }
#pragma unroll
        for (int mt = 0; mt < 2; ++mt) { // A: K quarters split lane-halves / upper VGPRs
            int m = (wm * 2 + mt) * 16 + rl;
            int base = m * BK + khalf * 8;
            a[mt].q[0] = *(const v4u*)&lA[base];        // K = khalf*8 .. +7
            a[mt].q[1] = *(const v4u*)&lA[base + 16];   // K = 16 + khalf*8 .. +7
        }
#pragma unroll
        for (int mt = 0; mt < 2; ++mt)
#pragma unroll
            for (int nt = 0; nt < 4; ++nt)
                acc[mt][nt] = __builtin_amdgcn_wmma_f32_16x16x32_bf16(
                    false, a[mt].bf, false, b[nt].bf, (short)0, acc[mt][nt], false, false);

        if (i + 1 < nIter && (!A_F32 || !B_F32)) wait_async();
        __syncthreads();
    }

    // ---- epilogue: C layout — lane = N within tile; VGPR r -> M = r + 8*(lane>=16) ----
    float*          Cf = (float*)Cg + (size_t)bz * strideC;
    unsigned short* Ch = (unsigned short*)Cg + (size_t)bz * strideC;
    unsigned short* Xp = (MODE >= 4) ? (auxPtr + (size_t)bz * strideAux) : nullptr;
    const unsigned nl  = lane & 15;
    const unsigned mhi = (lane >> 4) * 8;
#pragma unroll
    for (int mt = 0; mt < 2; ++mt) {
        const unsigned mgbase = m0 + (wm * 2 + mt) * 16 + mhi;   // 8 consecutive rows r=0..7
#pragma unroll
        for (int nt = 0; nt < 4; ++nt) {
            const unsigned ng = n0 + (wn * 4 + nt) * 16 + nl;
            const v8f& A8 = acc[mt][nt];
            if (MODE == 0) {
#pragma unroll
                for (int r = 0; r < 8; ++r)
                    Cf[(mgbase + r) * (unsigned)ldo + ng] = A8[r];
            } else if (MODE == 1 || MODE == 5) {
#pragma unroll
                for (int r = 0; r < 8; ++r)
                    Ch[(mgbase + r) * (unsigned)ldo + ng] = f2bf(A8[r]);
            }
            if (MODE == 2) {                       // packed transposed b128 store
                v8bf h;
#pragma unroll
                for (int r = 0; r < 8; ++r) h[r] = (__bf16)A8[r];   // 4x v_cvt_pk_bf16_f32
                *(v8bf*)&Ch[ng * (unsigned)ldo + mgbase] = h;
            } else if (MODE == 4) {                // packed transposed store + residual
                v8bf av = *(const v8bf*)&Xp[ng * (unsigned)ldadd + mgbase];
                v8bf h;
#pragma unroll
                for (int r = 0; r < 8; ++r) h[r] = (__bf16)(A8[r] + (float)av[r]);
                *(v8bf*)&Ch[ng * (unsigned)ldo + mgbase] = h;
            } else if (MODE == 5) {                // dual: packed transposed copy to aux
                v8bf h;
#pragma unroll
                for (int r = 0; r < 8; ++r) h[r] = (__bf16)A8[r];
                *(v8bf*)&Xp[ng * (unsigned)ldadd + mgbase] = h;
            }
        }
    }
}

// softmax over the batch axis (8 entries per position), bf16 in / bf16 out, elementwise
__global__ __launch_bounds__(256) void softmax_batch(
    const unsigned short* __restrict__ sc, unsigned short* __restrict__ aff, unsigned SS)
{
    unsigned i = blockIdx.x * 256 + threadIdx.x;
    if (i >= SS) return;
    float e[8];
    float m = -3.0e38f;
#pragma unroll
    for (int n = 0; n < 8; ++n) { e[n] = bf2f(sc[n * SS + i]); m = fmaxf(m, e[n]); }
    float s = 0.f;
#pragma unroll
    for (int n = 0; n < 8; ++n) { e[n] = __expf(e[n] - m); s += e[n]; }
    float inv = 1.f / s;
#pragma unroll
    for (int n = 0; n < 8; ++n) aff[n * SS + i] = f2bf(e[n] * inv);
}

extern "C" void kernel_launch(void* const* d_in, const int* in_sizes, int n_in,
                              void* d_out, int out_size, void* d_ws, size_t ws_size,
                              hipStream_t stream) {
    const float* x    = (const float*)d_in[0];   // [8][256][2304]
    const float* Wk   = (const float*)d_in[1];   // [128][256]
    const float* Wq   = (const float*)d_in[2];   // [128][256]
    const float* Wv   = (const float*)d_in[3];   // [128][256]
    const float* Wout = (const float*)d_in[4];   // [256][128]
    float* y = (float*)d_out;                    // [8][256][2304]

    const int NB = 8, CH = 256, S = 2304, HD = 128;
    const long long SS = (long long)S * S;

    // workspace (bf16 elements); all big-GEMM B-operands kept pre-transposed
    unsigned short* ws   = (unsigned short*)d_ws;
    unsigned short* qT   = ws;                          // [8][2304][128]
    unsigned short* kT   = qT   + (size_t)NB * S * HD;  // [8][2304][128]
    unsigned short* v    = kT   + (size_t)NB * S * HD;  // [8][128][2304]
    unsigned short* vT   = v    + (size_t)NB * HD * S;  // [8][2304][128]
    unsigned short* opvT = vT   + (size_t)NB * S * HD;  // [8][2304][128]
    unsigned short* scT  = opvT + (size_t)NB * S * HD;  // [8][2304][2304]
    unsigned short* affT = scT  + (size_t)NB * SS;      // [8][2304][2304]

    dim3 blk(256);

    // qT[n] = (Wq * x[n])^T : [s][hid]
    wmma_gemm<true, true, 2><<<dim3(S / BN, HD / BM, NB), blk, 0, stream>>>(
        Wq, x, qT, nullptr, CH, CH, S, HD, 0,
        0, (long long)CH * S, (long long)S * HD, 0);
    // kT[n] = (Wk * x[n])^T : [s][hid]
    wmma_gemm<true, true, 2><<<dim3(S / BN, HD / BM, NB), blk, 0, stream>>>(
        Wk, x, kT, nullptr, CH, CH, S, HD, 0,
        0, (long long)CH * S, (long long)S * HD, 0);
    // v[n] = Wv * x[n] : [hid][s], plus vT[n] : [s][hid] (dual store)
    wmma_gemm<true, true, 5><<<dim3(S / BN, HD / BM, NB), blk, 0, stream>>>(
        Wv, x, v, vT, CH, CH, S, S, HD,
        0, (long long)CH * S, (long long)HD * S, (long long)S * HD);
    // scoresT[n] = qT[n] * kT[n]^T : [t][s]   (A=qT [t][hid], BT=kT [s][hid])
    wmma_gemm<false, false, 1><<<dim3(S / BN, S / BM, NB), blk, 0, stream>>>(
        qT, kT, scT, nullptr, HD, HD, HD, S, 0,
        (long long)S * HD, (long long)S * HD, SS, 0);
    // softmax across batch axis n (elementwise in (s,t), layout-agnostic)
    softmax_batch<<<dim3((unsigned)(SS / 256)), blk, 0, stream>>>(scT, affT, (unsigned)SS);
    // opvT[n] = (v[n]*aff[n] + v[n])^T : [t][hid]  (A=v [hid][s], BT=affT [t][s], res=vT)
    wmma_gemm<false, false, 4><<<dim3(S / BN, HD / BM, NB), blk, 0, stream>>>(
        v, affT, opvT, vT, S, S, S, HD, HD,
        (long long)HD * S, SS, (long long)S * HD, (long long)S * HD);
    // y[n] = Wout * (out+v)[n] : [c][s]   (A=Wout fp32 [256][128], BT=opvT [s][hid])
    wmma_gemm<true, false, 0><<<dim3(S / BN, CH / BM, NB), blk, 0, stream>>>(
        Wout, opvT, y, nullptr, HD, HD, HD, S, 0,
        0, (long long)S * HD, (long long)CH * S, 0);
}
